// MultiGATConvStack_4913442586731
// MI455X (gfx1250) — matrix operand order repaired and checked by reading, not compiled, not run on hardware
//
#include <hip/hip_runtime.h>

// ---------------- problem constants (from the reference) ----------------
#define N1V     20000
#define N2V     10000
#define NEDGE   150000
#define NHEAD   3
#define HCV     192          // HEADS*NSTATE
#define NLAY    5
#define FMK     4096         // 64*64
#define FMKB    4160         // FMK + 64 bias rows
#define FMSTEPS (FMKB / 32)  // 130

typedef _Float16 f16_t;
typedef f16_t v16h __attribute__((ext_vector_type(16)));
typedef float v8f  __attribute__((ext_vector_type(8)));

// monotone float<->uint mapping for atomicMax-based segment max
__device__ __forceinline__ unsigned f2ord(float x) {
  unsigned u = __float_as_uint(x);
  return (u & 0x80000000u) ? ~u : (u | 0x80000000u);
}
__device__ __forceinline__ float ord2f(unsigned u) {
  return __uint_as_float((u & 0x80000000u) ? (u & 0x7fffffffu) : ~u);
}

__device__ __forceinline__ v16h scale16(float fv, const float4& c0, const float4& c1,
                                        const float4& c2, const float4& c3) {
  v16h a;
  a[0]=(f16_t)(fv*c0.x);  a[1]=(f16_t)(fv*c0.y);  a[2]=(f16_t)(fv*c0.z);  a[3]=(f16_t)(fv*c0.w);
  a[4]=(f16_t)(fv*c1.x);  a[5]=(f16_t)(fv*c1.y);  a[6]=(f16_t)(fv*c1.z);  a[7]=(f16_t)(fv*c1.w);
  a[8]=(f16_t)(fv*c2.x);  a[9]=(f16_t)(fv*c2.y);  a[10]=(f16_t)(fv*c2.z); a[11]=(f16_t)(fv*c2.w);
  a[12]=(f16_t)(fv*c3.x); a[13]=(f16_t)(fv*c3.y); a[14]=(f16_t)(fv*c3.z); a[15]=(f16_t)(fv*c3.w);
  return a;
}

// ---------------------------------------------------------------------------
// Pack an f32 row-major weight [K x N] (ld=ldb) into WMMA B-fragment order:
// out[((nt*ksteps + ks)*32 + lane)*16 + e] = B[ks*32 + (e&7) + 16*(e>>3) + 8*(lane>>4)][nt*16 + lane&15]
// zero-padded outside K/N so the GEMM needs no guards.
// ---------------------------------------------------------------------------
__global__ void pack_b_ker(const float* __restrict__ B, int K, int N, int ldb,
                           int ksteps, f16_t* __restrict__ out, int total)
{
  int i = blockIdx.x * blockDim.x + threadIdx.x;
  if (i >= total) return;
  int e    = i & 15;
  int lane = (i >> 4) & 31;
  int ks   = (i >> 9) % ksteps;
  int nt   = i / (ksteps * 512);
  int kgb  = ks * 32 + (e & 7) + ((e >> 3) << 4) + ((lane >> 4) << 3);
  int col  = nt * 16 + (lane & 15);
  float v = (kgb < K && col < N) ? B[(size_t)kgb * ldb + col] : 0.f;
  out[i] = (f16_t)v;
}

// ---------------------------------------------------------------------------
// One-wave 16x16-tile GEMM, compile-time K,N:
//   C[M,N] = A[M,K](f32, row-major) * Bpacked + (bias) (relu) (accum) (mask)
// M = gridDim.x*16 (20000/10000 are multiples of 16).
// ---------------------------------------------------------------------------
template<int K, int N>
__global__ __launch_bounds__(32)
void gemm16_wmma(const float* __restrict__ A, int lda,
                 const f16_t* __restrict__ Bp,
                 const float* __restrict__ bias,
                 const float* __restrict__ Cin,
                 float* __restrict__ Cout, int ldc, int flags, // 1=relu 2=accum
                 const int* __restrict__ maskTypes, int maskVal)
{
  constexpr int KSTEPS = (K + 31) / 32;
  const int lane = threadIdx.x;
  const int m16  = lane & 15;
  const int hlf  = lane >> 4;
  const int rowBase = blockIdx.x * 16;
  const int colBase = blockIdx.y * 16;
  const float* arow = A + (size_t)(rowBase + m16) * lda;
  const f16_t* bptr = Bp + ((size_t)blockIdx.y * KSTEPS * 32 + lane) * 16;

  v8f acc = {};
#pragma unroll
  for (int ks = 0; ks < KSTEPS; ++ks) {
    const int kb = ks * 32;
    float4 c0 = make_float4(0.f, 0.f, 0.f, 0.f), c1 = c0, c2 = c0, c3 = c0;
    if constexpr (K == 8) {
      if (hlf == 0) {                       // only K=0..7 exists
        c0 = *(const float4*)(arow + 0);
        c1 = *(const float4*)(arow + 4);
      }
    } else {
      // A layout: lane holds K-chunks [kb+8h, +8) and [kb+16+8h, +8)
      c0 = *(const float4*)(arow + kb + 8 * hlf);
      c1 = *(const float4*)(arow + kb + 8 * hlf + 4);
      c2 = *(const float4*)(arow + kb + 16 + 8 * hlf);
      c3 = *(const float4*)(arow + kb + 16 + 8 * hlf + 4);
    }
    v16h a = scale16(1.f, c0, c1, c2, c3);
    v16h b = *(const v16h*)(bptr + (size_t)ks * 512);
    acc = __builtin_amdgcn_wmma_f32_16x16x32_f16(false, a, false, b, (short)0, acc,
                                                 false, false);
  }

#pragma unroll
  for (int r = 0; r < 8; ++r) {
    int row = rowBase + r + (hlf << 3);
    int col = colBase + m16;
    bool colOK = true;
    if constexpr ((N & 15) != 0) colOK = (col < N);
    if (colOK) {
      float v = acc[r];
      if (flags & 2) v += Cin[row * (size_t)ldc + col];
      if (bias)      v += bias[col];
      if (flags & 1) v = fmaxf(v, 0.f);
      if (!maskTypes || maskTypes[row] == maskVal)
        Cout[row * (size_t)ldc + col] = v;
    }
  }
}

// ---------------------------------------------------------------------------
// Factor-net fused GEMM:
//   out[n,i] = sum_{k,j} f1[n,k]*tmp[n,j]*W[k, i*64+j] + sum_j b[i*64+j]*tmp[n,j]
// A[n, k*64+j] = f1[n,k]*tmp[n,j]. Key structure: within the main K range the
// tmp-chunk of the A-fragment depends only on the j-half, so the lane's 32 tmp
// values are hoisted into registers once; the k-loop only rescales+converts.
// B3p pre-packed f16 fragments [4 coltiles][130 ksteps][32 lanes][16].
// Masked store: rows with factor_types[n]==tval.
// ---------------------------------------------------------------------------
__global__ __launch_bounds__(32)
void fm_gemm(const float* __restrict__ f1, const float* __restrict__ tmp,
             const f16_t* __restrict__ B3p,
             const int* __restrict__ types, int tval,
             float* __restrict__ out)
{
  __shared__ float sf[16 * 64];
  __shared__ float st[16 * 64];
  const int lane = threadIdx.x;
  const int tile = blockIdx.x;
  for (int i = lane; i < 16 * 64; i += 32) {
    int r = i >> 6, c = i & 63;
    sf[i] = f1[(tile * 16 + r) * 64 + c];
    st[i] = tmp[(tile * 16 + r) * 64 + c];
  }
  __syncthreads();

  const int m16 = lane & 15;
  const int hlf = lane >> 4;
  const f16_t* blane = B3p + (size_t)lane * 16;

  // hoist the lane's tmp-row fragment chunks: jh=0 -> t[0..3], jh=1 -> t[4..7]
  const float* srow = &st[m16 * 64 + 8 * hlf];
  float4 t0 = *(const float4*)(srow);      float4 t1 = *(const float4*)(srow + 4);
  float4 t2 = *(const float4*)(srow + 16); float4 t3 = *(const float4*)(srow + 20);
  float4 t4 = *(const float4*)(srow + 32); float4 t5 = *(const float4*)(srow + 36);
  float4 t6 = *(const float4*)(srow + 48); float4 t7 = *(const float4*)(srow + 52);

  v8f acc[4] = {};

  for (int k4 = 0; k4 < 16; ++k4) {          // k advances 4 per iteration
    float4 f4 = *(const float4*)&sf[m16 * 64 + 4 * k4];
#pragma unroll
    for (int kk = 0; kk < 4; ++kk) {
      int k = 4 * k4 + kk;
      float fv = (kk == 0) ? f4.x : (kk == 1) ? f4.y : (kk == 2) ? f4.z : f4.w;
#pragma unroll
      for (int jh = 0; jh < 2; ++jh) {       // ks = 2*k + jh
        v16h a = jh ? scale16(fv, t4, t5, t6, t7) : scale16(fv, t0, t1, t2, t3);
        int ks = 2 * k + jh;
#pragma unroll
        for (int ib = 0; ib < 4; ++ib) {
          v16h b = *(const v16h*)(blane + ((size_t)ib * FMSTEPS + ks) * 512);
          acc[ib] = __builtin_amdgcn_wmma_f32_16x16x32_f16(false, a, false, b, (short)0,
                                                           acc[ib], false, false);
        }
      }
    }
  }

  // bias tail: ks = 128 (t[0..3]) and 129 (t[4..7]), scale 1
#pragma unroll
  for (int jh = 0; jh < 2; ++jh) {
    v16h a = jh ? scale16(1.f, t4, t5, t6, t7) : scale16(1.f, t0, t1, t2, t3);
    int ks = 128 + jh;
#pragma unroll
    for (int ib = 0; ib < 4; ++ib) {
      v16h b = *(const v16h*)(blane + ((size_t)ib * FMSTEPS + ks) * 512);
      acc[ib] = __builtin_amdgcn_wmma_f32_16x16x32_f16(false, a, false, b, (short)0,
                                                       acc[ib], false, false);
    }
  }

#pragma unroll
  for (int r = 0; r < 8; ++r) {
    int node = tile * 16 + r + (hlf << 3);
    if (types[node] == tval) {
#pragma unroll
      for (int ib = 0; ib < 4; ++ib)
        out[node * 64 + ib * 16 + m16] = acc[ib][r];
    }
  }
}

// Build packed factor-net B3 fragments directly:
// logical B3[row=(k*64+j)][i] = fnetW[k][i*64+j]; rows 4096..4159 = bias matrix.
__global__ void build_b3_ker(const float* __restrict__ fnetW,
                             const float* __restrict__ fnetb,
                             f16_t* __restrict__ B3p)
{
  int i = blockIdx.x * blockDim.x + threadIdx.x;
  if (i >= 4 * FMSTEPS * 512) return;
  int e    = i & 15;
  int lane = (i >> 4) & 31;
  int ks   = (i >> 9) % FMSTEPS;
  int ib   = i / (FMSTEPS * 512);
  int kgb  = ks * 32 + (e & 7) + ((e >> 3) << 4) + ((lane >> 4) << 3);
  int col  = ib * 16 + (lane & 15);
  float v;
  if (kgb < FMK) {
    int k = kgb >> 6, j = kgb & 63;
    v = fnetW[k * FMK + col * 64 + j];
  } else {
    v = fnetb[col * 64 + (kgb - FMK)];
  }
  B3p[i] = (f16_t)v;
}

// wdeff[k][h] = sum_c Wd[k, h*64+c] * ad[h*64+c]   (so a_d = x_dst @ wdeff)
__global__ void wdeff_ker(const float* __restrict__ Wd, const float* __restrict__ ad,
                          float* __restrict__ wdeff)
{
  int i = blockIdx.x * blockDim.x + threadIdx.x;
  if (i >= 64 * NHEAD) return;
  int k = i / NHEAD, h = i % NHEAD;
  float s = 0.f;
  for (int c = 0; c < 64; ++c) s += Wd[k * HCV + h * 64 + c] * ad[h * 64 + c];
  wdeff[i] = s;
}

// a_d[n,h] = sum_k x[n,k]*wdeff[k,h]
__global__ void adst_ker(const float* __restrict__ x, const float* __restrict__ wdeff,
                         float* __restrict__ ad_out, int total)
{
  int i = blockIdx.x * blockDim.x + threadIdx.x;
  if (i >= total) return;
  int n = i / NHEAD, h = i % NHEAD;
  float s = 0.f;
  for (int k = 0; k < 64; ++k) s += x[n * 64 + k] * wdeff[k * NHEAD + h];
  ad_out[i] = s;
}

// a_s[n,h] = sum_c hs[n, h*64+c]*asv[h*64+c]
__global__ void asrc_ker(const float* __restrict__ hs, const float* __restrict__ asv,
                         float* __restrict__ as_out, int total)
{
  int i = blockIdx.x * blockDim.x + threadIdx.x;
  if (i >= total) return;
  int n = i / NHEAD, h = i % NHEAD;
  float s = 0.f;
  for (int c = 0; c < 64; ++c) s += hs[n * HCV + h * 64 + c] * asv[h * 64 + c];
  as_out[i] = s;
}

// pass 1: e = leakyrelu(a_s[src]+a_d[dst], 0.2); segment max via ordered atomicMax
__global__ void edge_max_ker(const int* __restrict__ src, const int* __restrict__ dst,
                             const float* __restrict__ a_s, const float* __restrict__ a_d,
                             float* __restrict__ ebuf, unsigned* __restrict__ emax)
{
  int i = blockIdx.x * blockDim.x + threadIdx.x;
  if (i >= NEDGE * NHEAD) return;
  int e = i / NHEAD, h = i % NHEAD;
  int s = src[e], d = dst[e];
  float v = a_s[s * NHEAD + h] + a_d[d * NHEAD + h];
  v = (v >= 0.f) ? v : 0.2f * v;
  ebuf[i] = v;
  atomicMax(&emax[d * NHEAD + h], f2ord(v));
}

// pass 2: ex = exp(e - emax[dst]); denom[dst] += ex
__global__ void edge_exp_ker(const int* __restrict__ dst,
                             float* __restrict__ ebuf, const unsigned* __restrict__ emax,
                             float* __restrict__ denom)
{
  int i = blockIdx.x * blockDim.x + threadIdx.x;
  if (i >= NEDGE * NHEAD) return;
  int e = i / NHEAD, h = i % NHEAD;
  int d = dst[e];
  float ex = expf(ebuf[i] - ord2f(emax[d * NHEAD + h]));
  ebuf[i] = ex;
  atomicAdd(&denom[d * NHEAD + h], ex);
}

// pass 3: out[dst, m] += hs[src, m] * ex/denom[dst,h]   (block per edge, 192 threads)
__global__ void edge_scatter_ker(const int* __restrict__ src, const int* __restrict__ dst,
                                 const float* __restrict__ hs, const float* __restrict__ ebuf,
                                 const float* __restrict__ denom, float* __restrict__ out)
{
  int e = blockIdx.x;
  int m = threadIdx.x;            // 0..191, h = m>>6
  int s = src[e], d = dst[e];
  int h = m >> 6;
  float w = ebuf[e * NHEAD + h] / denom[d * NHEAD + h];
  atomicAdd(&out[(size_t)d * HCV + m], hs[(size_t)s * HCV + m] * w);
}

// pooled[n,s] = sum_{t<3} (out[n, 3s+t] + b[3s+t])  (reference's reshape semantics)
__global__ void pool_ker(const float* __restrict__ gout, const float* __restrict__ b,
                         float* __restrict__ dst, int total)
{
  int i = blockIdx.x * blockDim.x + threadIdx.x;
  if (i >= total) return;
  int n = i >> 6, s = i & 63;
  const float* row = gout + (size_t)n * HCV + 3 * s;
  dst[i] = row[0] + row[1] + row[2] + b[3 * s] + b[3 * s + 1] + b[3 * s + 2];
}

// ---------------------------------------------------------------------------
extern "C" void kernel_launch(void* const* d_in, const int* in_sizes, int n_in,
                              void* d_out, int out_size, void* d_ws, size_t ws_size,
                              hipStream_t stream)
{
  (void)in_sizes; (void)n_in; (void)out_size; (void)ws_size;

  // inputs, setup_inputs() dict order (params flattened in insertion order)
  const float* x1     = (const float*)d_in[0];
  const float* x2     = (const float*)d_in[1];
  const int*   esrc   = (const int*)  d_in[2];
  const int*   edst   = (const int*)  d_in[3];
  const int*   ftypes = (const int*)  d_in[4];
  const float* e1W1 = (const float*)d_in[5];  const float* e1b1 = (const float*)d_in[6];
  const float* e1W2 = (const float*)d_in[7];  const float* e1b2 = (const float*)d_in[8];
  const float* e1W3 = (const float*)d_in[9];  const float* e1b3 = (const float*)d_in[10];
  const float* e2W1 = (const float*)d_in[11]; const float* e2b1 = (const float*)d_in[12];
  const float* e2W2 = (const float*)d_in[13]; const float* e2b2 = (const float*)d_in[14];
  const float* e2W3 = (const float*)d_in[15]; const float* e2b3 = (const float*)d_in[16];
  const float* fnetW = (const float*)d_in[17];
  const float* fnetb = (const float*)d_in[18];
  const float* p12Ws = (const float*)d_in[19]; const float* p12Wd = (const float*)d_in[20];
  const float* p12as = (const float*)d_in[21]; const float* p12ad = (const float*)d_in[22];
  const float* p12b  = (const float*)d_in[23];
  const float* p21Ws = (const float*)d_in[24]; const float* p21Wd = (const float*)d_in[25];
  const float* p21as = (const float*)d_in[26]; const float* p21ad = (const float*)d_in[27];
  const float* p21b  = (const float*)d_in[28];
  const float* dW1 = (const float*)d_in[29]; const float* db1 = (const float*)d_in[30];
  const float* dW2 = (const float*)d_in[31]; const float* db2 = (const float*)d_in[32];
  const float* dW3 = (const float*)d_in[33]; const float* db3 = (const float*)d_in[34];

  // workspace carve-up (256B aligned)
  size_t off = 0;
  auto carve = [&](size_t bytes) {
    void* p = (char*)d_ws + off;
    off = (off + bytes + 255) & ~(size_t)255;
    return p;
  };
  float* f1    = (float*)carve((size_t)N1V * 64 * 4);
  float* h1a   = (float*)carve((size_t)N1V * 64 * 4);
  float* h1b   = (float*)carve((size_t)N1V * 64 * 4);
  float* h2a   = (float*)carve((size_t)N2V * 64 * 4);
  float* h2b   = (float*)carve((size_t)N2V * 64 * 4);
  float* hsbuf = (float*)carve((size_t)N1V * HCV * 4);
  float* outb  = (float*)carve((size_t)N1V * HCV * 4);
  float* tmpp  = (float*)carve((size_t)N1V * 64 * 4);
  float* bufA  = (float*)carve((size_t)N1V * 64 * 4);
  float* bufB  = (float*)carve((size_t)N1V * 64 * 4);
  float*    a_s   = (float*)   carve((size_t)N1V * NHEAD * 4);
  float*    a_d   = (float*)   carve((size_t)N1V * NHEAD * 4);
  unsigned* emaxb = (unsigned*)carve((size_t)N1V * NHEAD * 4);
  float*    denom = (float*)   carve((size_t)N1V * NHEAD * 4);
  float*    ebuf  = (float*)   carve((size_t)NEDGE * NHEAD * 4);
  float*    wdeff = (float*)   carve(64 * NHEAD * 4);
  f16_t*    B3p   = (f16_t*)   carve((size_t)4 * FMSTEPS * 512 * 2);

  auto packB = [&](const float* B, int K, int N, int ldb) -> f16_t* {
    int ntiles = (N + 15) / 16, ksteps = (K + 31) / 32;
    int total = ntiles * ksteps * 512;
    f16_t* p = (f16_t*)carve((size_t)total * 2);
    pack_b_ker<<<(total + 255) / 256, 256, 0, stream>>>(B, K, N, ldb, ksteps, p, total);
    return p;
  };

  // ---- pre-pack all weights into WMMA fragment order (once per launch) ----
  build_b3_ker<<<(4 * FMSTEPS * 512 + 255) / 256, 256, 0, stream>>>(fnetW, fnetb, B3p);
  f16_t* pe2W1 = packB(e2W1, 8, 64, 64);
  f16_t* pe2W2 = packB(e2W2, 64, 64, 64);
  f16_t* pe2W3 = packB(e2W3, 64, 64, 64);
  f16_t *pe1W1[2], *pe1W2[2], *pe1W3[2];
  for (int t = 0; t < 2; ++t) {
    pe1W1[t] = packB(e1W1 + (size_t)t * 8 * 64,  8,  64, 64);
    pe1W2[t] = packB(e1W2 + (size_t)t * 64 * 64, 64, 64, 64);
    pe1W3[t] = packB(e1W3 + (size_t)t * 64 * 64, 64, 64, 64);
  }
  f16_t *pp12hi[NLAY], *pp12lo[NLAY];
  for (int l = 0; l < NLAY; ++l) {
    pp12hi[l] = packB(p12Ws + (size_t)l * 128 * HCV,            64, HCV, HCV);
    pp12lo[l] = packB(p12Ws + (size_t)l * 128 * HCV + 64 * HCV, 64, HCV, HCV);
  }
  f16_t* pp21[2 * NLAY];
  for (int q = 0; q < 2 * NLAY; ++q)
    pp21[q] = packB(p21Ws + (size_t)q * 64 * HCV, 64, HCV, HCV);
  f16_t* pdW1 = packB(dW1, 64, 64, 64);
  f16_t* pdW2 = packB(dW2, 64, 64, 64);
  f16_t* pdW3 = packB(dW3, 64, 4, 4);

  const int RELU = 1, ACC = 2;
  const dim3 g1_64(N1V / 16, 4),  g1_192(N1V / 16, 12);
  const dim3 g2_64(N2V / 16, 4),  g2_192(N2V / 16, 12);

  // ---- enc2: x2 -> h2a ----
  gemm16_wmma<8,  64><<<g2_64, 32, 0, stream>>>(x2,   8,  pe2W1, e2b1, nullptr, bufA, 64, RELU, nullptr, 0);
  gemm16_wmma<64, 64><<<g2_64, 32, 0, stream>>>(bufA, 64, pe2W2, e2b2, nullptr, bufB, 64, RELU, nullptr, 0);
  gemm16_wmma<64, 64><<<g2_64, 32, 0, stream>>>(bufB, 64, pe2W3, e2b3, nullptr, h2a,  64, 0,    nullptr, 0);

  // ---- enc1 per type, masked select into f1 ----
  for (int t = 0; t < 2; ++t) {
    gemm16_wmma<8,  64><<<g1_64, 32, 0, stream>>>(x1,   8,  pe1W1[t], e1b1 + t * 64, nullptr, bufA, 64, RELU, nullptr, 0);
    gemm16_wmma<64, 64><<<g1_64, 32, 0, stream>>>(bufA, 64, pe1W2[t], e1b2 + t * 64, nullptr, bufB, 64, RELU, nullptr, 0);
    gemm16_wmma<64, 64><<<g1_64, 32, 0, stream>>>(bufB, 64, pe1W3[t], e1b3 + t * 64, nullptr, f1,   64, 0,    ftypes, t);
  }
  hipMemcpyAsync(h1a, f1, (size_t)N1V * 64 * 4, hipMemcpyDeviceToDevice, stream);

  float *h1 = h1a, *h1n = h1b, *h2 = h2a, *h2n = h2b;

  for (int l = 0; l < NLAY; ++l) {
    // ======== p12 GAT: factors(h1e=[h1,f1]) -> variables ========
    gemm16_wmma<64, HCV><<<g1_192, 32, 0, stream>>>(h1, 64, pp12hi[l], nullptr, nullptr, hsbuf, HCV, 0,   nullptr, 0);
    gemm16_wmma<64, HCV><<<g1_192, 32, 0, stream>>>(f1, 64, pp12lo[l], nullptr, hsbuf,   hsbuf, HCV, ACC, nullptr, 0);
    wdeff_ker<<<1, 192, 0, stream>>>(p12Wd + (size_t)l * 64 * HCV, p12ad + l * HCV, wdeff);
    asrc_ker<<<(N1V * NHEAD + 255) / 256, 256, 0, stream>>>(hsbuf, p12as + l * HCV, a_s, N1V * NHEAD);
    adst_ker<<<(N2V * NHEAD + 255) / 256, 256, 0, stream>>>(h2, wdeff, a_d, N2V * NHEAD);
    hipMemsetAsync(emaxb, 0, (size_t)N2V * NHEAD * 4, stream);
    hipMemsetAsync(denom, 0, (size_t)N2V * NHEAD * 4, stream);
    hipMemsetAsync(outb,  0, (size_t)N2V * HCV * 4,  stream);
    edge_max_ker<<<(NEDGE * NHEAD + 255) / 256, 256, 0, stream>>>(esrc, edst, a_s, a_d, ebuf, emaxb);
    edge_exp_ker<<<(NEDGE * NHEAD + 255) / 256, 256, 0, stream>>>(edst, ebuf, emaxb, denom);
    edge_scatter_ker<<<NEDGE, HCV, 0, stream>>>(esrc, edst, hsbuf, ebuf, denom, outb);
    pool_ker<<<(N2V * 64 + 255) / 256, 256, 0, stream>>>(outb, p12b + l * HCV, h2n, N2V * 64);

    // ======== p21 GAT per type: variables(h2) -> factors, + factor-net ========
    for (int t = 0; t < 2; ++t) {
      size_t pw = (size_t)t * NLAY + l;
      gemm16_wmma<64, HCV><<<g2_192, 32, 0, stream>>>(h2, 64, pp21[pw], nullptr, nullptr, hsbuf, HCV, 0, nullptr, 0);
      wdeff_ker<<<1, 192, 0, stream>>>(p21Wd + pw * 64 * HCV, p21ad + pw * HCV, wdeff);
      asrc_ker<<<(N2V * NHEAD + 255) / 256, 256, 0, stream>>>(hsbuf, p21as + pw * HCV, a_s, N2V * NHEAD);
      adst_ker<<<(N1V * NHEAD + 255) / 256, 256, 0, stream>>>(h1, wdeff, a_d, N1V * NHEAD);
      hipMemsetAsync(emaxb, 0, (size_t)N1V * NHEAD * 4, stream);
      hipMemsetAsync(denom, 0, (size_t)N1V * NHEAD * 4, stream);
      hipMemsetAsync(outb,  0, (size_t)N1V * HCV * 4,  stream);
      // roles swapped: source nodes are variables (edge_dst), dst are factors (edge_src)
      edge_max_ker<<<(NEDGE * NHEAD + 255) / 256, 256, 0, stream>>>(edst, esrc, a_s, a_d, ebuf, emaxb);
      edge_exp_ker<<<(NEDGE * NHEAD + 255) / 256, 256, 0, stream>>>(esrc, ebuf, emaxb, denom);
      edge_scatter_ker<<<NEDGE, HCV, 0, stream>>>(edst, esrc, hsbuf, ebuf, denom, outb);
      pool_ker<<<(N1V * 64 + 255) / 256, 256, 0, stream>>>(outb, p21b + pw * HCV, tmpp, N1V * 64);
      // fused factor-net batched matvec, masked store into h1n
      fm_gemm<<<N1V / 16, 32, 0, stream>>>(f1, tmpp, B3p, ftypes, t, h1n);
    }

    { float* s = h1; h1 = h1n; h1n = s; }
    { float* s = h2; h2 = h2n; h2n = s; }
  }

  // ---- decoder: h2 -> d_out [N2 x 4] ----
  gemm16_wmma<64, 64><<<g2_64, 32, 0, stream>>>(h2,   64, pdW1, db1, nullptr, bufA, 64, RELU, nullptr, 0);
  gemm16_wmma<64, 64><<<g2_64, 32, 0, stream>>>(bufA, 64, pdW2, db2, nullptr, bufB, 64, RELU, nullptr, 0);
  gemm16_wmma<64, 4><<<dim3(N2V / 16, 1), 32, 0, stream>>>(bufB, 64, pdW3, db3, nullptr,
                                                           (float*)d_out, 4, 0, nullptr, 0);
}
